// HyperDimensionalSpikeCoding_48550310314474
// MI455X (gfx1250) — compile-verified, matrix operand order checked by reading, not run on hardware
//
#include <hip/hip_runtime.h>
#include <hip/hip_bf16.h>

#define H 10000
#define BATCHN 1024
#define IDIM 1024
#define NSLOTS 100
#define KPAD 10016   // 313 * 32
#define NPADS 112    // 7 * 16
#define EPSF 1e-8f
#define GEMM_MT 4    // M-tiles per wave (generic GEMM)
#define CIRC_MT 8    // M-tiles per wave (circulant GEMM)

typedef __bf16 bf16_t;
typedef __bf16 v16bf __attribute__((ext_vector_type(16)));
typedef float  v8f   __attribute__((ext_vector_type(8)));

// ------------------------------------------------------------- TDM support
#ifndef USE_TDM
#if defined(__has_builtin)
#if __has_builtin(__builtin_amdgcn_tensor_load_to_lds)
#define USE_TDM 1
#endif
#endif
#endif
#ifndef USE_TDM
#define USE_TDM 0
#endif

#if USE_TDM
typedef unsigned int u32x4 __attribute__((ext_vector_type(4)));
typedef int          i32x4 __attribute__((ext_vector_type(4)));
typedef int          i32x8 __attribute__((ext_vector_type(8)));

// 2D tile load: rows x rowlen bf16 elements, row stride = stride elems,
// landing contiguously at LDS byte offset lds_off.
__device__ __forceinline__ void tdm_load_2d(unsigned int lds_off, const void* gptr,
                                            unsigned int rowlen, unsigned int rows,
                                            unsigned int stride, unsigned int tdim0) {
  unsigned long long ga = (unsigned long long)gptr;
  union { unsigned int a[4]; u32x4 v; } g0;
  g0.a[0] = 1u;                                        // count=1 valid descriptor
  g0.a[1] = lds_off;                                   // lds_addr
  g0.a[2] = (unsigned int)(ga & 0xFFFFFFFFull);        // global_addr[31:0]
  g0.a[3] = (unsigned int)((ga >> 32) & 0x1FFFFFFull)  // global_addr[56:32]
          | (2u << 30);                                // type = 2 (image)
  union { int a[8]; i32x8 v; } g1;
  g1.a[0] = (int)(1u << 16);                                   // data_size = 2B
  g1.a[1] = (int)((tdim0 & 0xFFFFu) << 16);                    // tensor_dim0 lo
  g1.a[2] = (int)(((tdim0 >> 16) & 0xFFFFu) | (rows << 16));   // dim0 hi | tensor_dim1 lo
  g1.a[3] = (int)(rowlen << 16);                               // dim1 hi=0 | tile_dim0
  g1.a[4] = (int)rows;                                         // tile_dim1 | tile_dim2=0
  g1.a[5] = (int)stride;                                       // tensor_dim0_stride lo
  g1.a[6] = 0;                                                 // stride hi | dim1_stride lo
  g1.a[7] = 0;
  i32x4 z4 = {};
#if __clang_major__ >= 23
  i32x8 z8 = {};
  __builtin_amdgcn_tensor_load_to_lds(g0.v, g1.v, z4, z4, z8, 0);
#else
  __builtin_amdgcn_tensor_load_to_lds(g0.v, g1.v, z4, z4, 0);
#endif
}
#endif  // USE_TDM

// ---------------------------------------------------------------- utilities

__device__ __forceinline__ float rng01(unsigned int x) {
  x ^= 42u * 0x9E3779B9u;
  x ^= x >> 16; x *= 0x7FEB352Du;
  x ^= x >> 15; x *= 0x846CA68Bu;
  x ^= x >> 16;
  return (float)(x >> 8) * (1.0f / 16777216.0f);
}

__device__ __forceinline__ float block_reduce_sum(float v) {
  __shared__ float red[256];
  red[threadIdx.x] = v;
  __syncthreads();
  for (int s = 128; s > 0; s >>= 1) {
    if ((int)threadIdx.x < s) red[threadIdx.x] += red[threadIdx.x + s];
    __syncthreads();
  }
  return red[0];
}

__device__ __forceinline__ int slot_row(int s) { return (s == 0) ? (BATCHN - 1) : s; }

// ---------------------------------------------------------------- converts

__global__ void k_cvt_bf16(const float* __restrict__ src, bf16_t* __restrict__ dst, int n) {
  int i = blockIdx.x * blockDim.x + threadIdx.x;
  if (i < n) dst[i] = (bf16_t)src[i];
}

__global__ void k_cvt_pad(const float* __restrict__ src, bf16_t* __restrict__ dst, int rows) {
  int i = blockIdx.x * blockDim.x + threadIdx.x;
  int total = rows * KPAD;
  if (i >= total) return;
  int r = i / KPAD, c = i - r * KPAD;
  dst[i] = (c < H) ? (bf16_t)src[(size_t)r * H + c] : (bf16_t)0.0f;
}

__global__ void k_zero(float* __restrict__ dst, int n) {
  int i = blockIdx.x * blockDim.x + threadIdx.x;
  if (i < n) dst[i] = 0.0f;
}

// ---------------------------------------------------------------- binding fold

// out[n] = sum_j x[j] * y[(n - j) mod H]
__global__ void k_circ_conv(const float* __restrict__ x, const float* __restrict__ y,
                            float* __restrict__ out) {
  int n = blockIdx.x * blockDim.x + threadIdx.x;
  if (n >= H) return;
  float acc = 0.0f;
  int idx = n;
  for (int j = 0; j < H; ++j) {
    acc = fmaf(x[j], y[idx], acc);
    idx = (idx == 0) ? (H - 1) : (idx - 1);
  }
  out[n] = acc;
}

// hh[i] = h[i mod H], i in [0, 2H)
__global__ void k_build_hh(const float* __restrict__ h, bf16_t* __restrict__ hh) {
  int i = blockIdx.x * blockDim.x + threadIdx.x;
  if (i < 2 * H) hh[i] = (bf16_t)h[(i < H) ? i : (i - H)];
}

// ---------------------------------------------------------------- WMMA fragments

__device__ __forceinline__ v16bf load_a_frag(const bf16_t* __restrict__ A, int lda,
                                             int row, int kb, int lane) {
  int q0 = (lane < 16) ? 0 : 8;
  const bf16_t* p = A + (size_t)row * lda + kb + q0;
  union { float4 f[2]; v16bf v; } u;
  u.f[0] = *(const float4*)(p);
  u.f[1] = *(const float4*)(p + 16);
  return u.v;
}

__device__ __forceinline__ v16bf load_b_frag_lds(const unsigned short* __restrict__ sB,
                                                 int lane) {
  int n = lane & 15;
  int q0 = (lane < 16) ? 0 : 8;
  union { unsigned int u[8]; v16bf v; } r;
#pragma unroll
  for (int p = 0; p < 8; ++p) {
    int k = (p < 4) ? (q0 + 2 * p) : (16 + q0 + 2 * (p - 4));
    unsigned int lo = sB[k * 16 + n];
    unsigned int hi = sB[(k + 1) * 16 + n];
    r.u[p] = lo | (hi << 16);
  }
  return r.v;
}

// ---------------------------------------------------------------- generic bf16 WMMA GEMM
// 256 threads = 8 waves; each wave owns GEMM_MT 16x16 C tiles on one N tile.
// B tile staged to LDS by the Tensor Data Mover (fallback: coalesced b32 stores).

__global__ __launch_bounds__(256) void k_gemm_bf16(
    const bf16_t* __restrict__ A, const bf16_t* __restrict__ B, float* __restrict__ C,
    int M, int N, int K, int lda, int ldb, int ldc) {
  __shared__ unsigned short sB[32 * 16];
  int lane = threadIdx.x & 31;
  int wave = threadIdx.x >> 5;
  int n0 = blockIdx.x * 16;
  int mbase = blockIdx.y * (8 * GEMM_MT * 16) + wave * (GEMM_MT * 16);

  union Acc { v8f v; float f[8]; };
  Acc acc[GEMM_MT] = {};

  for (int kb = 0; kb < K; kb += 32) {
#if USE_TDM
    if (threadIdx.x < 32) {  // one wave issues the DMA; EXEC is ignored by TDM
      tdm_load_2d((unsigned int)(unsigned long long)&sB[0],
                  B + (size_t)kb * ldb + n0,
                  /*rowlen=*/16u, /*rows=*/32u, (unsigned)ldb, (unsigned)(N - n0));
      __builtin_amdgcn_s_wait_tensorcnt(0);
    }
#else
    {
      int t = threadIdx.x;
      int r = t >> 3, d = t & 7;
      const unsigned int* src = (const unsigned int*)(B + (size_t)(kb + r) * ldb + n0);
      ((unsigned int*)sB)[t] = src[d];
    }
#endif
    __syncthreads();
    v16bf bfrag = load_b_frag_lds(sB, lane);
#pragma unroll
    for (int t = 0; t < GEMM_MT; ++t) {
      int row = mbase + t * 16 + (lane & 15);
      v16bf afrag = load_a_frag(A, lda, row, kb, lane);
      acc[t].v = __builtin_amdgcn_wmma_f32_16x16x32_bf16(
          false, afrag, false, bfrag, (short)0, acc[t].v, false, false);
    }
    __syncthreads();
  }

  int col = n0 + (lane & 15);
  if (col < N) {
    int roff = (lane < 16) ? 0 : 8;
#pragma unroll
    for (int t = 0; t < GEMM_MT; ++t) {
      int rbase = mbase + t * 16 + roff;
#pragma unroll
      for (int r = 0; r < 8; ++r)
        C[(size_t)(rbase + r) * ldc + col] = acc[t].f[r];
    }
  }
}

// ---------------------------------------------------------------- circulant GEMM (dominant)
// bound = hyper @ Circ(h); B fragments synthesized from a pre-packed LDS table:
// sP[i] = hh[i-16] | hh[i-17]<<16  ->  each fragment dword is ONE ds_load_b32.
// 8 M-tiles/wave -> 1 ds_load_b32 + 2 global_load_b128 per v_wmma.

__global__ __launch_bounds__(256) void k_gemm_circ(
    const bf16_t* __restrict__ A, const bf16_t* __restrict__ hh,
    float* __restrict__ spikes, float* __restrict__ cleaned) {
  __shared__ unsigned int sP[2 * H + 32];  // 78.25 KB of the 320 KB WGP LDS
  const unsigned short* hhu = (const unsigned short*)hh;
  for (int i = threadIdx.x; i < 2 * H + 32; i += 256) {
    int j = i - 16;
    unsigned int lo = (j >= 0 && j < 2 * H) ? (unsigned int)hhu[j] : 0u;
    unsigned int hi = (j >= 1 && j - 1 < 2 * H) ? (unsigned int)hhu[j - 1] : 0u;
    sP[i] = lo | (hi << 16);
  }
  __syncthreads();

  int lane = threadIdx.x & 31;
  int wave = threadIdx.x >> 5;
  int n0 = blockIdx.x * 16;
  int mbase = blockIdx.y * (8 * CIRC_MT * 16) + wave * (CIRC_MT * 16);
  int nlane = n0 + (lane & 15);
  int q0 = (lane < 16) ? 0 : 8;

  union Acc { v8f v; float f[8]; };
  Acc acc[CIRC_MT] = {};

#pragma unroll
  for (int t = 0; t < CIRC_MT; ++t)
    __builtin_prefetch(A + (size_t)(mbase + t * 16 + (lane & 15)) * KPAD, 0, 3);

  for (int kb = 0; kb < KPAD; kb += 32) {
    union { unsigned int u[8]; v16bf v; } bf;
#pragma unroll
    for (int p = 0; p < 8; ++p) {
      int k = (p < 4) ? (q0 + 2 * p) : (16 + q0 + 2 * (p - 4));
      bf.u[p] = sP[16 + H + nlane - (kb + k)];  // index in [1, 20015]
    }
#pragma unroll
    for (int t = 0; t < CIRC_MT; ++t) {
      int row = mbase + t * 16 + (lane & 15);
      v16bf afrag = load_a_frag(A, KPAD, row, kb, lane);
      acc[t].v = __builtin_amdgcn_wmma_f32_16x16x32_bf16(
          false, afrag, false, bf.v, (short)0, acc[t].v, false, false);
    }
  }

  int roff = (lane < 16) ? 0 : 8;
#pragma unroll
  for (int t = 0; t < CIRC_MT; ++t) {
#pragma unroll
    for (int r = 0; r < 8; ++r) {
      int row = mbase + t * 16 + roff + r;
      float x = acc[t].f[r];
      float p = 1.0f / (1.0f + __expf(-x));
      float u = rng01((unsigned int)row * (unsigned int)H + (unsigned int)nlane);
      float s = (u < p) ? 1.0f : 0.0f;
      spikes[(size_t)row * H + nlane] = s;
      cleaned[(size_t)row * H + nlane] = s;
    }
  }
}

// ---------------------------------------------------------------- cleanup helpers

__global__ __launch_bounds__(256) void k_mem_norms(const float* __restrict__ spikes,
                                                   float* __restrict__ mem_norm) {
  int s = blockIdx.x;
  int row = slot_row(s);
  float acc = 0.0f;
  for (int j = threadIdx.x; j < H; j += 256) acc += spikes[(size_t)row * H + j];
  float tot = block_reduce_sum(acc);
  if (threadIdx.x == 0) mem_norm[s] = fmaxf(sqrtf(tot), EPSF);
}

__global__ __launch_bounds__(256) void k_row_norms(const float* __restrict__ X,
                                                   float* __restrict__ out) {
  int row = blockIdx.x;
  float acc = 0.0f;
  for (int j = threadIdx.x; j < H; j += 256) {
    float v = X[(size_t)row * H + j];
    acc += v * v;
  }
  float tot = block_reduce_sum(acc);
  if (threadIdx.x == 0) out[row] = fmaxf(sqrtf(tot), EPSF);
}

__global__ void k_build_memT(const float* __restrict__ spikes, bf16_t* __restrict__ memT) {
  int i = blockIdx.x * blockDim.x + threadIdx.x;
  int total = KPAD * NPADS;
  if (i >= total) return;
  int k = i / NPADS, n = i - k * NPADS;
  float v = 0.0f;
  if (n < NSLOTS && k < H) v = spikes[(size_t)slot_row(n) * H + k];
  memT[i] = (bf16_t)v;
}

__global__ __launch_bounds__(256) void k_blend(
    const float* __restrict__ sims, const float* __restrict__ c_norm,
    const float* __restrict__ mem_norm, const float* __restrict__ spikes,
    float* __restrict__ cleaned, const float* __restrict__ thr_p) {
  __shared__ float sw[256];
  __shared__ int si[256];
  int row = blockIdx.x;
  int t = threadIdx.x;
  float w = -1e30f;
  if (t < NSLOTS) w = sims[(size_t)row * NPADS + t] / (c_norm[row] * mem_norm[t]);
  sw[t] = w; si[t] = t;
  __syncthreads();
  for (int s = 128; s > 0; s >>= 1) {
    if (t < s && sw[t + s] > sw[t]) { sw[t] = sw[t + s]; si[t] = si[t + s]; }
    __syncthreads();
  }
  float wbest = sw[0];
  int best = si[0];
  float thr = *thr_p;
  if (wbest > thr) {
    int mrow = slot_row(best);
    for (int j = t; j < H; j += 256) {
      float c = cleaned[(size_t)row * H + j];
      float m = spikes[(size_t)mrow * H + j];
      cleaned[(size_t)row * H + j] = (1.0f - wbest) * c + wbest * m;
    }
  }
}

__global__ __launch_bounds__(256) void k_binding_stats(const float* __restrict__ spikes,
                                                       float* __restrict__ out) {
  int row = blockIdx.x;
  float acc = 0.0f;
  for (int j = threadIdx.x; j < H; j += 256) acc += spikes[(size_t)row * H + j];
  float s1 = block_reduce_sum(acc);
  if (threadIdx.x == 0) {
    float p = s1 / (float)H;
    float var = (s1 - s1 * s1 / (float)H) / (float)(H - 1);
    float d = p - 0.1f;
    out[row] = __expf(-d * d) * var;
  }
}

// ---------------------------------------------------------------- launch

extern "C" void kernel_launch(void* const* d_in, const int* in_sizes, int n_in,
                              void* d_out, int out_size, void* d_ws, size_t ws_size,
                              hipStream_t stream) {
  const float* spikes_in = (const float*)d_in[0];  // 1024 x 1024
  const float* proj      = (const float*)d_in[1];  // 1024 x 10000
  const float* bops      = (const float*)d_in[2];  // 8 x 10000
  const float* thr       = (const float*)d_in[3];  // scalar

  float* out_spikes = (float*)d_out;
  float* out_res    = out_spikes + (size_t)BATCHN * H;
  float* out_clean  = out_res + (size_t)BATCHN * NSLOTS;
  float* out_bs     = out_clean + (size_t)BATCHN * H;

  char* ws = (char*)d_ws;
  auto alloc = [&](size_t bytes) {
    char* p = ws;
    ws += (bytes + 255) & ~(size_t)255;
    return p;
  };
  float*  hyper_f32 = (float*)alloc(sizeof(float) * BATCHN * H);
  bf16_t* hyper_bf  = (bf16_t*)alloc(sizeof(bf16_t) * BATCHN * KPAD);
  bf16_t* proj_bf   = (bf16_t*)alloc(sizeof(bf16_t) * IDIM * H);
  bf16_t* spk_bf    = (bf16_t*)alloc(sizeof(bf16_t) * BATCHN * IDIM);
  bf16_t* clean_bf  = (bf16_t*)alloc(sizeof(bf16_t) * BATCHN * KPAD);
  bf16_t* memT_bf   = (bf16_t*)alloc(sizeof(bf16_t) * KPAD * NPADS);
  bf16_t* hh_bf     = (bf16_t*)alloc(sizeof(bf16_t) * 2 * H);
  float*  t0        = (float*)alloc(sizeof(float) * H);
  float*  t1        = (float*)alloc(sizeof(float) * H);
  float*  mem_norm  = (float*)alloc(sizeof(float) * NSLOTS);
  float*  c_norm    = (float*)alloc(sizeof(float) * BATCHN);
  float*  sims      = (float*)alloc(sizeof(float) * BATCHN * NPADS);

  dim3 blk(256);

  // Stage A: hyper = spikes @ proj  (bf16 WMMA, TDM-staged B tiles)
  k_cvt_bf16<<<(BATCHN * IDIM + 255) / 256, blk, 0, stream>>>(spikes_in, spk_bf, BATCHN * IDIM);
  k_cvt_bf16<<<(IDIM * H + 255) / 256, blk, 0, stream>>>(proj, proj_bf, IDIM * H);
  k_gemm_bf16<<<dim3(H / 16, BATCHN / 512), blk, 0, stream>>>(
      spk_bf, proj_bf, hyper_f32, BATCHN, H, IDIM, IDIM, H, H);
  k_cvt_pad<<<(BATCHN * KPAD + 255) / 256, blk, 0, stream>>>(hyper_f32, hyper_bf, BATCHN);

  // Stage B: fold 8 binding ops into one kernel h, then circulant GEMM
  int cb = (H + 255) / 256;
  k_circ_conv<<<cb, blk, 0, stream>>>(bops + 0 * H, bops + 1 * H, t0);
  k_circ_conv<<<cb, blk, 0, stream>>>(t0, bops + 2 * H, t1);
  k_circ_conv<<<cb, blk, 0, stream>>>(t1, bops + 3 * H, t0);
  k_circ_conv<<<cb, blk, 0, stream>>>(t0, bops + 4 * H, t1);
  k_circ_conv<<<cb, blk, 0, stream>>>(t1, bops + 5 * H, t0);
  k_circ_conv<<<cb, blk, 0, stream>>>(t0, bops + 6 * H, t1);
  k_circ_conv<<<cb, blk, 0, stream>>>(t1, bops + 7 * H, t0);
  k_build_hh<<<(2 * H + 255) / 256, blk, 0, stream>>>(t0, hh_bf);
  k_gemm_circ<<<dim3(H / 16, BATCHN / (8 * CIRC_MT * 16)), blk, 0, stream>>>(
      hyper_bf, hh_bf, out_spikes, out_clean);

  // memory_resonance output is all zeros
  k_zero<<<(BATCHN * NSLOTS + 255) / 256, blk, 0, stream>>>(out_res, BATCHN * NSLOTS);

  // Stage C: memory + cleanup x3
  k_mem_norms<<<NSLOTS, blk, 0, stream>>>(out_spikes, mem_norm);
  k_build_memT<<<(KPAD * NPADS + 255) / 256, blk, 0, stream>>>(out_spikes, memT_bf);
  for (int it = 0; it < 3; ++it) {
    k_row_norms<<<BATCHN, blk, 0, stream>>>(out_clean, c_norm);
    k_cvt_pad<<<(BATCHN * KPAD + 255) / 256, blk, 0, stream>>>(out_clean, clean_bf, BATCHN);
    k_gemm_bf16<<<dim3(NPADS / 16, BATCHN / 512), blk, 0, stream>>>(
        clean_bf, memT_bf, sims, BATCHN, NPADS, KPAD, KPAD, NPADS, NPADS);
    k_blend<<<BATCHN, blk, 0, stream>>>(sims, c_norm, mem_norm, out_spikes, out_clean, thr);
  }

  // Stage D: binding strength
  k_binding_stats<<<BATCHN, blk, 0, stream>>>(out_spikes, out_bs);

  (void)in_sizes; (void)n_in; (void)out_size; (void)ws_size;
}
